// SinkhornKnopp_59897613910349
// MI455X (gfx1250) — matrix operand order repaired and checked by reading, not compiled, not run on hardware
//
#include <hip/hip_runtime.h>
#include <math.h>

// ---------------- problem constants ----------------
#define NROWS   12288   // 4096 features + 8192 queue
#define NFEAT   4096
#define DDIM    256
#define KOUT    1000
#define KP      1008    // K padded to 63 * 16
#define KTILES  63
#define MTILES  768     // 12288 / 16
#define MGROUPS 192     // 768 / 4  (each wave does 4 M-tiles)
#define JCHUNKS 48      // 12288 / 256
#define INV_EPS 20.0f   // 1 / 0.05

typedef __attribute__((ext_vector_type(16))) _Float16 v16h;
typedef __attribute__((ext_vector_type(8)))  _Float16 v8h;
typedef __attribute__((ext_vector_type(8)))  float    v8f;

union HPack { v16h v; v8h h8[2]; };

// ---------------- reduction helper ----------------
__device__ inline float blockReduceSum(float v, float* s) {
    int t = threadIdx.x;
    #pragma unroll
    for (int o = 16; o > 0; o >>= 1) v += __shfl_down(v, o, 32);
    int wid = t >> 5, lane = t & 31;
    if (lane == 0) s[wid] = v;
    __syncthreads();
    float r = (t < 8) ? s[t] : 0.0f;
    if (t < 32) {
        #pragma unroll
        for (int o = 4; o > 0; o >>= 1) r += __shfl_down(r, o, 32);
    }
    __syncthreads();
    if (t == 0) s[0] = r;
    __syncthreads();
    return s[0];
}

// ---------------- 1: L2-normalize feats rows -> f16 ----------------
__global__ __launch_bounds__(256) void norm_feats_kernel(
    const float* __restrict__ features, const float* __restrict__ queue,
    _Float16* __restrict__ fn) {
    __shared__ float s[32];
    int row = blockIdx.x, t = threadIdx.x;
    const float* src = (row < NFEAT) ? (features + (size_t)row * DDIM)
                                     : (queue + (size_t)(row - NFEAT) * DDIM);
    float x = src[t];
    float total = blockReduceSum(x * x, s);
    float scale = 1.0f / fmaxf(sqrtf(total), 1e-12f);
    fn[(size_t)row * DDIM + t] = (_Float16)(x * scale);
}

// ---------------- 2: L2-normalize head rows (len 1000), store transposed+padded ----------------
__global__ __launch_bounds__(256) void norm_head_kernel(
    const float* __restrict__ head, _Float16* __restrict__ hnT) {
    __shared__ float s[32];
    int d = blockIdx.x, t = threadIdx.x;
    float acc = 0.0f;
    for (int k = t; k < KOUT; k += 256) {
        float x = head[(size_t)d * KOUT + k];
        acc += x * x;
    }
    float total = blockReduceSum(acc, s);
    float scale = 1.0f / fmaxf(sqrtf(total), 1e-12f);
    for (int k = t; k < KP; k += 256) {
        float v = (k < KOUT) ? head[(size_t)d * KOUT + k] * scale : 0.0f;
        hnT[(size_t)k * DDIM + d] = (_Float16)v;
    }
}

// ---------------- 3: WMMA GEMM (fn @ hnT^T) + exp -> E[12288][1008] f32 ----------------
// One wave per 64x16 macro-tile (4 M-tiles x 1 N-tile): B fragment reused 4x,
// 4 independent accumulators -> 32 back-to-back v_wmma with no RAW hazards.
__global__ __launch_bounds__(256) void gemm_exp_kernel(
    const _Float16* __restrict__ fn, const _Float16* __restrict__ hnT,
    float* __restrict__ E) {
    int wid  = blockIdx.x * 8 + (threadIdx.x >> 5);   // global wave id
    int mg   = wid / KTILES;                          // 0..191 (M group of 4 tiles)
    int nt   = wid % KTILES;                          // 0..62
    int lane = threadIdx.x & 31;
    int half_ = lane >> 4;       // selects K = 16*half_ .. 16*half_+15 (contiguous order)
    int lrow  = lane & 15;

    const _Float16* abase = fn  + (size_t)(mg * 64 + lrow) * DDIM;  // A: M = lrow (+16*m)
    const _Float16* brow  = hnT + (size_t)(nt * 16 + lrow) * DDIM;  // B: N = lrow

    v8f acc[4] = {v8f{}, v8f{}, v8f{}, v8f{}};
    #pragma unroll
    for (int kb = 0; kb < DDIM; kb += 32) {
        // B 32x16 f16 layout: lane half h holds K = h*16 .. h*16+15 (contiguous in hnT row)
        HPack b;
        b.v = *(const v16h*)(brow + kb + half_ * 16);
        #pragma unroll
        for (int m = 0; m < 4; ++m) {
            // A 16x32 f16 layout: lane half h holds K = h*16 .. h*16+15 (matches B)
            const _Float16* arow = abase + (size_t)m * 16 * DDIM;
            HPack a;
            a.h8[0] = *(const v8h*)(arow + kb + half_ * 16);
            a.h8[1] = *(const v8h*)(arow + kb + half_ * 16 + 8);
            acc[m] = __builtin_amdgcn_wmma_f32_16x16x32_f16(
                false, a.v, false, b.v, (short)0, acc[m], false, false);
        }
    }

    // C layout: VGPR v -> M = v + 8*half, N = lrow
    int n = nt * 16 + lrow;
    bool valid = (n < KOUT);
    #pragma unroll
    for (int m = 0; m < 4; ++m) {
        int mbase = mg * 64 + m * 16 + half_ * 8;
        #pragma unroll
        for (int v = 0; v < 8; ++v) {
            float e = valid ? __expf(acc[m][v] * INV_EPS) : 0.0f;  // padded cols -> 0
            E[(size_t)(mbase + v) * KP + n] = e;
        }
    }
}

// ---------------- 4: init col scale c = 1 ----------------
__global__ void init_c_kernel(float* __restrict__ c) {
    int j = blockIdx.x * 256 + threadIdx.x;
    if (j < NROWS) c[j] = 1.0f;
}

// ---------------- 5a: partial column sums of E weighted by c ----------------
// part[jc][k] = sum_{j in 256-chunk jc} E[j][k] * c[j]
__global__ __launch_bounds__(256) void col_partial_kernel(
    const float* __restrict__ E, const float* __restrict__ c,
    float* __restrict__ part) {
    int kt = blockIdx.x;             // 0..62
    int jc = blockIdx.y;             // 0..47
    int t  = threadIdx.x;
    int j  = jc * 256 + t;
    float cj = c[j];
    const float* row = E + (size_t)j * KP + kt * 16;
    float acc[16];
    #pragma unroll
    for (int i = 0; i < 16; ++i) acc[i] = row[i] * cj;
    #pragma unroll
    for (int i = 0; i < 16; ++i) {
        #pragma unroll
        for (int o = 16; o > 0; o >>= 1) acc[i] += __shfl_down(acc[i], o, 32);
    }
    __shared__ float s[16 * 8];
    int wid = t >> 5, lane = t & 31;
    if (lane == 0) {
        #pragma unroll
        for (int i = 0; i < 16; ++i) s[i * 8 + wid] = acc[i];
    }
    __syncthreads();
    if (t < 16) {
        float v = 0.0f;
        #pragma unroll
        for (int w = 0; w < 8; ++w) v += s[t * 8 + w];
        part[(size_t)jc * KP + kt * 16 + t] = v;
    }
}

// ---------------- 5b: finalize r[k] = 1/(K * colsum_k) ----------------
__global__ void finalize_r_kernel(const float* __restrict__ part, float* __restrict__ r) {
    int k = blockIdx.x * 256 + threadIdx.x;
    if (k >= KP) return;
    float s = 0.0f;
    #pragma unroll 4
    for (int jc = 0; jc < JCHUNKS; ++jc) s += part[(size_t)jc * KP + k];
    r[k] = (k < KOUT) ? 1.0f / ((float)KOUT * s) : 0.0f;
}

// ---------------- 5c: c[j] = 1/(B * sum_k E[j][k]*r[k]) ----------------
__global__ __launch_bounds__(256) void update_c_kernel(
    const float* __restrict__ E, const float* __restrict__ r,
    float* __restrict__ c) {
    __shared__ float s[32];
    int j = blockIdx.x, t = threadIdx.x;
    const float* row = E + (size_t)j * KP;
    float acc = 0.0f;
    for (int k = t; k < KP; k += 256) acc += row[k] * r[k];
    float total = blockReduceSum(acc, s);
    if (t == 0) c[j] = 1.0f / ((float)NROWS * total);
}

// ---------------- 6: out[n][k] = B * E[n][k] * r[k] * c[n], n < 4096 ----------------
__global__ __launch_bounds__(256) void write_out_kernel(
    const float* __restrict__ E, const float* __restrict__ r,
    const float* __restrict__ c, float* __restrict__ out) {
    int n = blockIdx.y;
    int k = blockIdx.x * 256 + threadIdx.x;
    if (k < KOUT)
        out[(size_t)n * KOUT + k] = (float)NROWS * E[(size_t)n * KP + k] * r[k] * c[n];
}

extern "C" void kernel_launch(void* const* d_in, const int* in_sizes, int n_in,
                              void* d_out, int out_size, void* d_ws, size_t ws_size,
                              hipStream_t stream) {
    const float* features = (const float*)d_in[0];  // [4096, 256]
    const float* head     = (const float*)d_in[1];  // [256, 1000]
    const float* queue    = (const float*)d_in[2];  // [8192, 256]
    float* out = (float*)d_out;                     // [4096, 1000]

    size_t off = 0;
    auto take = [&](size_t bytes) -> void* {
        void* p = (char*)d_ws + off;
        off += (bytes + 255) & ~(size_t)255;
        return p;
    };
    _Float16* fn   = (_Float16*)take((size_t)NROWS * DDIM * 2);   // 6.3 MB
    _Float16* hnT  = (_Float16*)take((size_t)KP * DDIM * 2);      // 0.5 MB
    float*    E    = (float*)take((size_t)NROWS * KP * 4);        // 49.5 MB (L2-resident)
    float*    part = (float*)take((size_t)JCHUNKS * KP * 4);
    float*    r    = (float*)take((size_t)KP * 4);
    float*    c    = (float*)take((size_t)NROWS * 4);

    norm_feats_kernel<<<NROWS, 256, 0, stream>>>(features, queue, fn);
    norm_head_kernel<<<DDIM, 256, 0, stream>>>(head, hnT);
    gemm_exp_kernel<<<(MGROUPS * KTILES + 7) / 8, 256, 0, stream>>>(fn, hnT, E);
    init_c_kernel<<<JCHUNKS, 256, 0, stream>>>(c);

    for (int it = 0; it < 3; ++it) {
        col_partial_kernel<<<dim3(KTILES, JCHUNKS), 256, 0, stream>>>(E, c, part);
        finalize_r_kernel<<<(KP + 255) / 256, 256, 0, stream>>>(part, r);
        update_c_kernel<<<NROWS, 256, 0, stream>>>(E, r, c);
    }
    write_out_kernel<<<dim3((KOUT + 255) / 256, NFEAT), 256, 0, stream>>>(E, r, c, out);
}
